// AttnCorrelation_12137577579281
// MI455X (gfx1250) — matrix operand order repaired and checked by reading, not compiled, not run on hardware
//
#include <hip/hip_runtime.h>

// ---------------- problem constants (from reference) ----------------
constexpr int B_  = 4;
constexpr int C_  = 256;
constexpr int H_  = 64;
constexpr int W_  = 128;
constexpr int D_  = 4;
constexpr int NS  = 9;          // shifts per axis
constexpr int NS2 = 81;         // total shifts
constexpr int Hp  = H_ + 2 * D_;  // 72
constexpr int Wp  = W_ + 2 * D_;  // 136
constexpr int CR  = 16;         // C / R  (R = 16)

typedef __attribute__((ext_vector_type(16))) __bf16 v16bf;
typedef __attribute__((ext_vector_type(8)))  __bf16 v8bf;
typedef __attribute__((ext_vector_type(8)))  float  v8f;

// ---------------- f32 -> bf16 conversion ----------------
__global__ void k_cvt_f1(const float* __restrict__ in, __bf16* __restrict__ out, int n) {
  for (int i = blockIdx.x * blockDim.x + threadIdx.x; i < n; i += gridDim.x * blockDim.x)
    out[i] = (__bf16)in[i];
}

// zero-padded bf16 copy of feat2: [B,C,Hp,Wp]
__global__ void k_cvt_f2pad(const float* __restrict__ in, __bf16* __restrict__ out) {
  const int n = B_ * C_ * Hp * Wp;
  for (int i = blockIdx.x * blockDim.x + threadIdx.x; i < n; i += gridDim.x * blockDim.x) {
    int wp = i % Wp;
    int t  = i / Wp;
    int hp = t % Hp;
    int bc = t / Hp;
    int h = hp - D_, w = wp - D_;
    float v = 0.f;
    if (h >= 0 && h < H_ && w >= 0 && w < W_) v = in[(bc * H_ + h) * W_ + w];
    out[i] = (__bf16)v;
  }
}

// ---------------- phase 1: 81-shift correlation via WMMA diagonal trick ----
// Block = 256 threads (8 waves). Block computes pooled[s, b, c0..c0+15].
// Each wave accumulates D = A(feat1 16ch x 32px) * B(feat2shift 32px x 16ch);
// diag(D) gives the per-channel correlation partial sums.
__global__ __launch_bounds__(256) void k_corr(const __bf16* __restrict__ f1,
                                              const __bf16* __restrict__ f2p,
                                              float* __restrict__ pooled) {
  __shared__ float sdiag[8][16];
  const int c0 = blockIdx.x * 16;
  const int s  = blockIdx.y;
  const int b  = blockIdx.z;
  const int di = s / NS, dj = s % NS;
  const int tid  = threadIdx.x;
  const int wave = tid >> 5;
  const int lane = tid & 31;
  const int hf   = lane >> 4;      // which K-half this lane covers
  const int ln   = lane & 15;      // A-row (M) == B-col (N) == channel-in-tile

  const long cch = (long)(b * C_ + c0 + ln);
  const __bf16* f1c  = f1  + cch * (long)(H_ * W_);
  const __bf16* f2pc = f2p + cch * (long)(Hp * Wp);

  v8f acc = {};
  // K = H*W = 8192 pixels; chunks of 32 pixels (never cross a row, W=128).
  // Wave w owns chunks w, w+8, ..., 32 chunks each.
  for (int t = 0; t < 32; ++t) {
    const int chunk = wave + (t << 3);
    const int p0 = chunk << 5;
    const int h  = p0 >> 7;          // p0 / W
    const int wb = p0 & (W_ - 1);    // p0 % W
    // A layout (16-bit 16x32): lane ln holds K = hf*16+[0..15]
    const __bf16* ap = f1c + (h * W_ + wb + hf * 16);
    v8bf a0 = *(const v8bf*)(ap);
    v8bf a1 = *(const v8bf*)(ap + 8);
    v16bf av;
#pragma unroll
    for (int e = 0; e < 8; ++e) { av[e] = a0[e]; av[e + 8] = a1[e]; }
    // B layout (32x16): lane holds K = hf*16+[0..15] for column N=ln.
    // Shifted feat2: row h+di, col (wb + K) + dj  (2-byte aligned, j varies)
    const __bf16* bp = f2pc + ((h + di) * Wp + wb + hf * 16 + dj);
    v16bf bv;
    __builtin_memcpy(&bv, bp, 32);
    acc = __builtin_amdgcn_wmma_f32_16x16x32_bf16(false, av, false, bv,
                                                  (short)0, acc, false, false);
  }
  // Diagonal extraction: VGPR r holds M = r + 8*hf, N = ln.
  // M==N lives in lanes 0..7 (r=lane) and 24..31 (r=lane-24).
  int m = -1, r = 0;
  if (lane < 8)        { m = lane;      r = lane;      }
  else if (lane >= 24) { m = lane - 16; r = lane - 24; }
  if (m >= 0) {
    float dv = 0.f;
#pragma unroll
    for (int rr = 0; rr < 8; ++rr) if (rr == r) dv = acc[rr];
    sdiag[wave][m] = dv;
  }
  __syncthreads();
  if (tid < 16) {
    float sum = 0.f;
#pragma unroll
    for (int w2 = 0; w2 < 8; ++w2) sum += sdiag[w2][tid];
    pooled[(s * B_ + b) * C_ + c0 + tid] = sum * (1.0f / (H_ * W_));
  }
}

// ---------------- phase 2: tiny MLP (256->16->256) + sigmoid ----------------
__global__ __launch_bounds__(256) void k_mlp(const float* __restrict__ pooled,
                                             const float* __restrict__ w1,
                                             const float* __restrict__ b1,
                                             const float* __restrict__ w2,
                                             const float* __restrict__ b2,
                                             float* __restrict__ g) {
  __shared__ float sp[C_];
  __shared__ float sa[CR];
  const int sb  = blockIdx.x;     // s*B + b
  const int tid = threadIdx.x;
  sp[tid] = pooled[sb * C_ + tid];
  __syncthreads();
  if (tid < CR) {
    float a = b1[tid];
    for (int c = 0; c < C_; ++c) a = fmaf(w1[tid * C_ + c], sp[c], a);
    sa[tid] = a;
  }
  __syncthreads();
  float a2 = b2[tid];
#pragma unroll
  for (int rr = 0; rr < CR; ++rr) a2 = fmaf(w2[tid * CR + rr], sa[rr], a2);
  g[sb * C_ + tid] = 1.0f / (1.0f + __expf(-a2));
}

// ---------------- zero-init output (d_out is poisoned by harness) ----------
__global__ void k_zero(float* __restrict__ p, int n) {
  int i = blockIdx.x * blockDim.x + threadIdx.x;
  if (i < n) p[i] = 0.f;
}

// ---------------- phase 3: gated channel-mean output ----------------------
// Tile 16x16 pixels, channel chunk CK=32 (keeps static LDS < 64KB).
// feat2p halo (24x24 per channel) + gate chunk staged in LDS; feat1 pixel
// values live in registers. 81 shifts reuse the halo. f32 atomics merge
// the 8 channel chunks.
constexpr int CK   = 32;
constexpr int TH   = 16, TW = 16;
constexpr int HALO = TH + 2 * D_;   // 24

__global__ __launch_bounds__(256) void k_out(const __bf16* __restrict__ f1,
                                             const __bf16* __restrict__ f2p,
                                             const float* __restrict__ g,
                                             float* __restrict__ out) {
  __shared__ __bf16 f2l[CK * HALO * HALO];   // 36864 B
  __shared__ float  gl[NS2 * CK];            // 10368 B
  const int wt = blockIdx.x;            // 0..7
  const int ht = blockIdx.y;            // 0..3
  const int bz = blockIdx.z;            // 0..31
  const int b  = bz >> 3;
  const int cc = (bz & 7) * CK;
  const int h0 = ht * TH, w0 = wt * TW;
  const int tid = threadIdx.x;
  const int tx = tid & 15, ty = tid >> 4;

  // stage gate chunk [s][c]
  for (int i = tid; i < NS2 * CK; i += 256) {
    int s = i / CK, c = i % CK;
    gl[i] = g[(s * B_ + b) * C_ + cc + c];
  }
  // stage feat2p halo [c][row][col] — tile is always interior of padded array
  for (int i = tid; i < CK * HALO * HALO; i += 256) {
    int c  = i / (HALO * HALO);
    int rr = (i % (HALO * HALO)) / HALO;
    int cl = i % HALO;
    f2l[i] = f2p[((long)(b * C_ + cc + c) * Hp + (h0 + rr)) * Wp + (w0 + cl)];
  }
  // per-thread feat1 pixel values in registers
  float f1r[CK];
#pragma unroll
  for (int c = 0; c < CK; ++c)
    f1r[c] = (float)f1[((long)(b * C_ + cc + c) * H_ + (h0 + ty)) * W_ + (w0 + tx)];
  __syncthreads();

  for (int s = 0; s < NS2; ++s) {
    const int di = s / NS, dj = s % NS;
    const __bf16* f2row = f2l + (ty + di) * HALO + (tx + dj);
    const float*  gs    = gl + s * CK;
    float acc = 0.f;
#pragma unroll
    for (int c = 0; c < CK; ++c)
      acc = fmaf(gs[c] * f1r[c], (float)f2row[c * (HALO * HALO)], acc);
    atomicAdd(out + ((long)(b * NS2 + s) * H_ + (h0 + ty)) * W_ + (w0 + tx),
              acc * (1.0f / C_));
  }
}

// ---------------- launcher ----------------
extern "C" void kernel_launch(void* const* d_in, const int* in_sizes, int n_in,
                              void* d_out, int out_size, void* d_ws, size_t ws_size,
                              hipStream_t stream) {
  (void)in_sizes; (void)n_in; (void)ws_size;
  const float* feat1 = (const float*)d_in[0];
  const float* feat2 = (const float*)d_in[1];
  const float* w1    = (const float*)d_in[2];
  const float* b1    = (const float*)d_in[3];
  const float* w2    = (const float*)d_in[4];
  const float* b2    = (const float*)d_in[5];
  float* out = (float*)d_out;

  const int n1  = B_ * C_ * H_ * W_;   // 8,388,608
  const int n2p = B_ * C_ * Hp * Wp;   // 10,027,008
  const int npl = NS2 * B_ * C_;       // 82,944

  char* ws = (char*)d_ws;
  size_t off = 0;
  auto take = [&](size_t bytes) {
    void* p = ws + off;
    off = (off + bytes + 255) & ~(size_t)255;
    return p;
  };
  __bf16* f1b    = (__bf16*)take((size_t)n1 * 2);
  __bf16* f2pb   = (__bf16*)take((size_t)n2p * 2);
  float*  pooled = (float*)take((size_t)npl * 4);
  float*  gate   = (float*)take((size_t)npl * 4);

  // 1) convert to bf16 (+ zero-pad feat2)
  k_cvt_f1<<<8192, 256, 0, stream>>>(feat1, f1b, n1);
  k_cvt_f2pad<<<8192, 256, 0, stream>>>(feat2, f2pb);

  // 2) WMMA correlation: pooled[s,b,c]
  k_corr<<<dim3(C_ / 16, NS2, B_), 256, 0, stream>>>(f1b, f2pb, pooled);

  // 3) MLP gate
  k_mlp<<<NS2 * B_, 256, 0, stream>>>(pooled, w1, b1, w2, b2, gate);

  // 4) zero out, then gated channel-mean accumulation
  k_zero<<<(out_size + 255) / 256, 256, 0, stream>>>(out, out_size);
  k_out<<<dim3(W_ / TW, H_ / TH, B_ * (C_ / CK)), 256, 0, stream>>>(f1b, f2pb, gate, out);
}
